// Attention_81149112090655
// MI455X (gfx1250) — matrix operand order repaired and checked by reading, not compiled, not run on hardware
//
#include <hip/hip_runtime.h>
#include <hip/hip_bf16.h>
#include <math.h>

#define DIMC   1152
#define HEADS  9
#define HD     128
#define MID    256
#define BB     8
#define NN     1024
#define TOK    (BB*NN)        // 8192 tokens
#define QKVN   (3*DIMC)       // 3456
#define UPN    (2*DIMC)       // 2304

typedef __attribute__((ext_vector_type(16))) __bf16 bf16x16;
typedef __attribute__((ext_vector_type(8)))  float  floatx8;

// ---------------- fragment loaders (wave32, 16x16x32 bf16 WMMA) ----------------

// A: 16x32 (MxK), row-major source with leading dim ld.
__device__ __forceinline__ bf16x16 load_frag_A(const __bf16* base, int ld) {
    int lane = threadIdx.x & 31;
    int m  = lane & 15;
    int kb = (lane >> 4) << 3;
    const __bf16* row = base + (size_t)m * ld;
    bf16x16 a;
#pragma unroll
    for (int v = 0; v < 8; ++v) {
        int k0 = ((v >> 2) << 4) + kb + ((v & 3) << 1);
        a[2*v]   = row[k0];
        a[2*v+1] = row[k0 + 1];
    }
    return a;
}

// Same as load_frag_A but fp32 source, converted on load (P tile in LDS).
__device__ __forceinline__ bf16x16 load_frag_A_f32(const float* base, int ld) {
    int lane = threadIdx.x & 31;
    int m  = lane & 15;
    int kb = (lane >> 4) << 3;
    const float* row = base + (size_t)m * ld;
    bf16x16 a;
#pragma unroll
    for (int v = 0; v < 8; ++v) {
        int k0 = ((v >> 2) << 4) + kb + ((v & 3) << 1);
        a[2*v]   = (__bf16)row[k0];
        a[2*v+1] = (__bf16)row[k0 + 1];
    }
    return a;
}

// B for C = A * W^T, W row-major [N,K]: B[k][n] = W[n][k]. Contiguous along k.
__device__ __forceinline__ bf16x16 load_frag_Bt(const __bf16* base, int ld) {
    int lane = threadIdx.x & 31;
    int n  = lane & 15;
    int kb = (lane >> 4) << 3;
    const __bf16* row = base + (size_t)n * ld;
    bf16x16 b;
#pragma unroll
    for (int v = 0; v < 8; ++v) {
        int k0 = ((v >> 2) << 4) + kb + ((v & 3) << 1);
        b[2*v]   = row[k0];
        b[2*v+1] = row[k0 + 1];
    }
    return b;
}

#define WMMA_BF16(a, b, c) \
    __builtin_amdgcn_wmma_f32_16x16x32_bf16(false, (a), false, (b), (short)0, (c), false, false)

// 64x64 tile micro-kernel: acc[4][4] += A(64xK) * W(64xK)^T
// 16 WMMAs per 8 fragment loads -> 4x operand reuse vs single-tile version.
__device__ __forceinline__ void gemm64_bt(const __bf16* __restrict__ A, int lda,
                                          const __bf16* __restrict__ B, int ldb,
                                          int K, floatx8 (&acc)[4][4]) {
    for (int k = 0; k < K; k += 32) {
        if (k + 32 < K) {                     // gfx1250 global_prefetch_b8
            __builtin_prefetch(A + k + 32, 0, 0);
            __builtin_prefetch(B + k + 32, 0, 0);
        }
        bf16x16 af[4], bf[4];
#pragma unroll
        for (int i = 0; i < 4; ++i) af[i] = load_frag_A (A + (size_t)(i*16)*lda + k, lda);
#pragma unroll
        for (int j = 0; j < 4; ++j) bf[j] = load_frag_Bt(B + (size_t)(j*16)*ldb + k, ldb);
#pragma unroll
        for (int i = 0; i < 4; ++i)
#pragma unroll
            for (int j = 0; j < 4; ++j)
                acc[i][j] = WMMA_BF16(af[i], bf[j], acc[i][j]);
    }
}

// ---------------- conversion / copy ----------------

__global__ void cvt_bf16_kernel(const float* __restrict__ in, __bf16* __restrict__ out, int n) {
    int i = blockIdx.x * 256 + threadIdx.x;
    if (i < n) out[i] = (__bf16)in[i];
}

__global__ void copy_f32_kernel(const float* __restrict__ in, float* __restrict__ out, int n) {
    int i = blockIdx.x * 256 + threadIdx.x;
    if (i < n) out[i] = in[i];
}

// ---------------- adapter hidden: T = tanh(x @ dw^T + db), bf16 out ----------------

__global__ __launch_bounds__(256)
void prefix_down_kernel(const __bf16* __restrict__ xb, const __bf16* __restrict__ dwb,
                        const float* __restrict__ db, __bf16* __restrict__ T,
                        const int* __restrict__ taskp, int need) {
    if (*taskp < need) return;
    int wave = (blockIdx.x * blockDim.x + threadIdx.x) >> 5;
    const int tiles_n = MID / 64;                 // 4
    int tm = wave / tiles_n, tn = wave % tiles_n;
    if (tm >= TOK / 64) return;

    floatx8 acc[4][4] = {};
    gemm64_bt(xb + (size_t)tm * 64 * DIMC, DIMC,
              dwb + (size_t)tn * 64 * DIMC, DIMC, DIMC, acc);

    int lane = threadIdx.x & 31;
    int n = lane & 15, mb = (lane >> 4) << 3;
    __bf16* out = T + (size_t)tm * 64 * MID + tn * 64;
#pragma unroll
    for (int i = 0; i < 4; ++i)
#pragma unroll
        for (int j = 0; j < 4; ++j) {
            float bias = db[tn * 64 + j * 16 + n];
#pragma unroll
            for (int v = 0; v < 8; ++v)
                out[(size_t)(i * 16 + mb + v) * MID + j * 16 + n] =
                    (__bf16)tanhf(acc[i][j][v] + bias);
        }
}

// ---------------- fused QKV: x@Wqkv^T (+ adapter up-projections on kv cols) ----------------

__global__ __launch_bounds__(256)
void qkv_kernel(const __bf16* __restrict__ xb, const __bf16* __restrict__ wqkvb,
                const __bf16* __restrict__ T1, const __bf16* __restrict__ upb,  const float* __restrict__ up_bias,
                const __bf16* __restrict__ T2, const __bf16* __restrict__ pupb, const float* __restrict__ pup_bias,
                __bf16* __restrict__ qkvb, const int* __restrict__ taskp) {
    int task = *taskp;
    int wave = (blockIdx.x * blockDim.x + threadIdx.x) >> 5;
    const int tiles_n = QKVN / 64;                // 54 (q: tiles 0..17, kv: 18..53)
    int tm = wave / tiles_n, tn = wave % tiles_n;
    if (tm >= TOK / 64) return;

    floatx8 acc[4][4] = {};
    gemm64_bt(xb + (size_t)tm * 64 * DIMC, DIMC,
              wqkvb + (size_t)tn * 64 * DIMC, DIMC, DIMC, acc);

    int lane = threadIdx.x & 31;
    int n = lane & 15, mb = (lane >> 4) << 3;

    if (tn >= DIMC / 64 && task >= 4) {           // kv columns: adapter contributions
        int kvcol = tn * 64 - DIMC;
        gemm64_bt(T1 + (size_t)tm * 64 * MID, MID,
                  upb + (size_t)kvcol * MID, MID, MID, acc);
#pragma unroll
        for (int j = 0; j < 4; ++j) {
            float bias = up_bias[kvcol + j * 16 + n];
#pragma unroll
            for (int i = 0; i < 4; ++i)
#pragma unroll
                for (int v = 0; v < 8; ++v) acc[i][j][v] += bias;
        }
        if (task >= 5) {
            gemm64_bt(T2 + (size_t)tm * 64 * MID, MID,
                      pupb + (size_t)kvcol * MID, MID, MID, acc);
#pragma unroll
            for (int j = 0; j < 4; ++j) {
                float bias = pup_bias[kvcol + j * 16 + n];
#pragma unroll
                for (int i = 0; i < 4; ++i)
#pragma unroll
                    for (int v = 0; v < 8; ++v) acc[i][j][v] += bias;
            }
        }
    }
    __bf16* out = qkvb + (size_t)tm * 64 * QKVN + tn * 64;
#pragma unroll
    for (int i = 0; i < 4; ++i)
#pragma unroll
        for (int j = 0; j < 4; ++j)
#pragma unroll
            for (int v = 0; v < 8; ++v)
                out[(size_t)(i * 16 + mb + v) * QKVN + j * 16 + n] = (__bf16)acc[i][j][v];
}

// ---------------- V transpose: Vt[b][h][d][key], coalesced both sides ----------------

__global__ __launch_bounds__(256)
void vtr_kernel(const __bf16* __restrict__ qkvb, __bf16* __restrict__ vt) {
    __shared__ __bf16 tile[16][17];
    int bid = blockIdx.x;                         // BB*HEADS*(NN/16)*(HD/16) blocks
    int dt = bid & 7;                             // HD/16
    int kt = (bid >> 3) & 63;                     // NN/16
    int h  = (bid >> 9) % HEADS;
    int b  = bid / (HEADS << 9);
    int tx = threadIdx.x & 15, ty = threadIdx.x >> 4;
    tile[ty][tx] = qkvb[(size_t)(b * NN + kt * 16 + ty) * QKVN + 2 * DIMC + h * HD + dt * 16 + tx];
    __syncthreads();
    vt[((size_t)(b * HEADS + h) * HD + dt * 16 + ty) * NN + kt * 16 + tx] = tile[tx][ty];
}

// ---------------- attention: one block per (b, h, 16-row tile); 8 waves ----------------

__global__ __launch_bounds__(256)
void attn_kernel(const __bf16* __restrict__ qkvb, const __bf16* __restrict__ vt,
                 __bf16* __restrict__ attnb) {
    __shared__ float S[16 * NN];                  // 64 KB score/prob tile

    int bid = blockIdx.x;
    int mt = bid % (NN / 16);
    int h  = (bid / (NN / 16)) % HEADS;
    int b  =  bid / (NN / 16) / HEADS;
    int wave = threadIdx.x >> 5;
    int lane = threadIdx.x & 31;
    const float scale = 0.08838834764831845f;     // 1/sqrt(128)

    // Phase 1: S = Q * K^T  (each wave owns 8 key tiles of 16; Q frags register-resident)
    const __bf16* qbase = qkvb + (size_t)(b * NN + mt * 16) * QKVN + h * HD;
    bf16x16 qa[4];
#pragma unroll
    for (int s = 0; s < 4; ++s) qa[s] = load_frag_A(qbase + s * 32, QKVN);

    const __bf16* kbase = qkvb + (size_t)(b * NN) * QKVN + DIMC + h * HD;
    for (int jt = wave; jt < NN / 16; jt += 8) {
        const __bf16* Bb = kbase + (size_t)jt * 16 * QKVN;
        floatx8 c = {};
#pragma unroll
        for (int s = 0; s < 4; ++s) {
            bf16x16 bf = load_frag_Bt(Bb + s * 32, QKVN);
            c = WMMA_BF16(qa[s], bf, c);
        }
        int n = lane & 15, mb = (lane >> 4) << 3;
#pragma unroll
        for (int v = 0; v < 8; ++v)
            S[(mb + v) * NN + jt * 16 + n] = c[v];
    }
    __syncthreads();

    // Phase 2: softmax normalized in place; each half-wave owns one row (shuffle reductions).
    {
        int r   = 2 * wave + (lane >> 4);
        int seg = lane & 15;
        float* Srow = S + r * NN;
        float mx = -1e30f;
        for (int j = seg; j < NN; j += 16) mx = fmaxf(mx, Srow[j]);
#pragma unroll
        for (int off = 1; off < 16; off <<= 1) mx = fmaxf(mx, __shfl_xor(mx, off, 32));
        float sum = 0.f;
        for (int j = seg; j < NN; j += 16) sum += __expf((Srow[j] - mx) * scale);
#pragma unroll
        for (int off = 1; off < 16; off <<= 1) sum += __shfl_xor(sum, off, 32);
        float inv = 1.0f / sum;
        for (int j = seg; j < NN; j += 16)
            Srow[j] = __expf((Srow[j] - mx) * scale) * inv;
    }
    __syncthreads();

    // Phase 3: O = P * V via transposed V (contiguous Bt frags); wave ct owns 16 output cols.
    int ct = wave;
    const __bf16* vtb = vt + ((size_t)(b * HEADS + h) * HD + ct * 16) * NN;
    floatx8 c = {};
    for (int ks = 0; ks < NN; ks += 32) {
        bf16x16 a  = load_frag_A_f32(S + ks, NN);
        bf16x16 bf = load_frag_Bt(vtb + ks, NN);
        c = WMMA_BF16(a, bf, c);
    }
    int n = lane & 15, mb = (lane >> 4) << 3;
    __bf16* out = attnb + (size_t)(b * NN + mt * 16) * DIMC + h * HD + ct * 16;
#pragma unroll
    for (int v = 0; v < 8; ++v)
        out[(size_t)(mb + v) * DIMC + n] = (__bf16)c[v];
}

// ---------------- output projection: out = attn @ Wproj^T + bproj (fp32 out) ----------------

__global__ __launch_bounds__(256)
void proj_kernel(const __bf16* __restrict__ attnb, const __bf16* __restrict__ wprojb,
                 const float* __restrict__ bproj, float* __restrict__ out) {
    int wave = (blockIdx.x * blockDim.x + threadIdx.x) >> 5;
    const int tiles_n = DIMC / 64;                // 18
    int tm = wave / tiles_n, tn = wave % tiles_n;
    if (tm >= TOK / 64) return;

    floatx8 acc[4][4] = {};
    gemm64_bt(attnb + (size_t)tm * 64 * DIMC, DIMC,
              wprojb + (size_t)tn * 64 * DIMC, DIMC, DIMC, acc);

    int lane = threadIdx.x & 31;
    int n = lane & 15, mb = (lane >> 4) << 3;
    float* o = out + (size_t)tm * 64 * DIMC + tn * 64;
#pragma unroll
    for (int i = 0; i < 4; ++i)
#pragma unroll
        for (int j = 0; j < 4; ++j) {
            float bias = bproj[tn * 64 + j * 16 + n];
#pragma unroll
            for (int v = 0; v < 8; ++v)
                o[(size_t)(i * 16 + mb + v) * DIMC + j * 16 + n] = acc[i][j][v] + bias;
        }
}

// ---------------- host launcher ----------------

extern "C" void kernel_launch(void* const* d_in, const int* in_sizes, int n_in,
                              void* d_out, int out_size, void* d_ws, size_t ws_size,
                              hipStream_t stream) {
    const float* x       = (const float*)d_in[0];
    const float* Wqkv    = (const float*)d_in[1];
    const float* Wproj   = (const float*)d_in[2];
    const float* bproj   = (const float*)d_in[3];
    const float* down_w  = (const float*)d_in[4];
    const float* down_b  = (const float*)d_in[5];
    const float* up_w    = (const float*)d_in[6];
    const float* up_b    = (const float*)d_in[7];
    const float* pdown_w = (const float*)d_in[8];
    const float* pdown_b = (const float*)d_in[9];
    const float* pup_w   = (const float*)d_in[10];
    const float* pup_b   = (const float*)d_in[11];
    const int*   taskp   = (const int*)d_in[12];
    float* out = (float*)d_out;

    // workspace carve (~136 MB)
    char* base = (char*)d_ws; size_t off = 0;
    auto carve = [&](size_t bytes) {
        void* q = base + off; off += (bytes + 255) & ~(size_t)255; return q;
    };
    __bf16* xb      = (__bf16*)carve((size_t)TOK * DIMC * 2);
    __bf16* wqkvb   = (__bf16*)carve((size_t)QKVN * DIMC * 2);
    __bf16* wprojb  = (__bf16*)carve((size_t)DIMC * DIMC * 2);
    __bf16* downb   = (__bf16*)carve((size_t)MID * DIMC * 2);
    __bf16* upb     = (__bf16*)carve((size_t)UPN * MID * 2);
    __bf16* pdownb  = (__bf16*)carve((size_t)MID * DIMC * 2);
    __bf16* pupb    = (__bf16*)carve((size_t)UPN * MID * 2);
    __bf16* T1      = (__bf16*)carve((size_t)TOK * MID * 2);
    __bf16* T2      = (__bf16*)carve((size_t)TOK * MID * 2);
    __bf16* qkvb    = (__bf16*)carve((size_t)TOK * QKVN * 2);
    __bf16* attnb   = (__bf16*)carve((size_t)TOK * DIMC * 2);
    __bf16* vt      = (__bf16*)carve((size_t)BB * HEADS * HD * NN * 2);
    (void)ws_size; (void)in_sizes; (void)n_in; (void)out_size;

    auto cvt = [&](const float* in, __bf16* o, int n) {
        cvt_bf16_kernel<<<dim3((n + 255) / 256), dim3(256), 0, stream>>>(in, o, n);
    };
    cvt(x,       xb,     TOK * DIMC);
    cvt(Wqkv,    wqkvb,  QKVN * DIMC);
    cvt(Wproj,   wprojb, DIMC * DIMC);
    cvt(down_w,  downb,  MID * DIMC);
    cvt(up_w,    upb,    UPN * MID);
    cvt(pdown_w, pdownb, MID * DIMC);
    cvt(pup_w,   pupb,   UPN * MID);

    // adapter hidden states (in-kernel task gating): down needs task>=4, pdown needs task>=5
    {
        int waves = (TOK / 64) * (MID / 64);      // 512
        dim3 grid(waves * 32 / 256);
        prefix_down_kernel<<<grid, dim3(256), 0, stream>>>(xb, downb,  down_b,  T1, taskp, 4);
        prefix_down_kernel<<<grid, dim3(256), 0, stream>>>(xb, pdownb, pdown_b, T2, taskp, 5);
    }

    // fused QKV
    {
        int waves = (TOK / 64) * (QKVN / 64);     // 6912
        qkv_kernel<<<dim3(waves * 32 / 256), dim3(256), 0, stream>>>(
            xb, wqkvb, T1, upb, up_b, T2, pupb, pup_b, qkvb, taskp);
    }

    // V transpose for P*V phase
    vtr_kernel<<<dim3(BB * HEADS * (NN / 16) * (HD / 16)), dim3(256), 0, stream>>>(qkvb, vt);

    // attention
    attn_kernel<<<dim3(BB * HEADS * (NN / 16)), dim3(256), 0, stream>>>(qkvb, vt, attnb);

    // projection -> fp32 output
    {
        int waves = (TOK / 64) * (DIMC / 64);     // 2304
        proj_kernel<<<dim3(waves * 32 / 256), dim3(256), 0, stream>>>(attnb, wprojb, bproj, out);
    }

    // second tuple element: down_w verbatim
    {
        int n = MID * DIMC;
        copy_f32_kernel<<<dim3((n + 255) / 256), dim3(256), 0, stream>>>(
            down_w, out + (size_t)TOK * DIMC, n);
    }
}
